// DynamicScaleConv2d_27410481283812
// MI455X (gfx1250) — matrix operand order repaired and checked by reading, not compiled, not run on hardware
//
#include <hip/hip_runtime.h>

// ---------------------------------------------------------------------------
// DynamicScaleConv2d on MI455X (gfx1250, wave32, WMMA).
//
// Roofline: core contraction = implicit im2col GEMM, M=B*H*W=147456,
// K=9*C=2304, N=O=256 -> 1.74e11 FLOP vs ~300MB minimal HBM traffic (13us
// @23.3TB/s) => matrix-pipe bound. Strategy:
//   * contraction on v_wmma_f32_16x16x32_f16 (f32 accumulate; N(0,1) data and
//     ~0.03-scale weights sit comfortably in f16 range),
//   * per block: 16 pixels x 256 outchans; the bilinearly sampled A-tile for a
//     whole tap (16x256 f16, 8KB) is built in LDS once per barrier and reused
//     by all 16 waves -> gather cost amortized 256x, only 9 barriers/block,
//   * each thread samples 8 consecutive channels: 4 corners = 8 coalesced
//     global_load_b128, one ds_store_b128; waves then run 8 back-to-back
//     WMMAs per tap with clause-able b128 fragment loads,
//   * weights pre-swizzled to the exact B-operand register image (two
//     contiguous global_load_b128 per fragment, 1.2MB L2-resident).
// ---------------------------------------------------------------------------

typedef __attribute__((ext_vector_type(16))) _Float16 v16h;
typedef __attribute__((ext_vector_type(8)))  _Float16 v8h;
typedef __attribute__((ext_vector_type(8)))  float    v8f;

#define B_ 4
#define C_ 256
#define O_ 256
#define H_ 192
#define W_ 192
#define HW_ (H_ * W_)
#define KC_ 72           // K chunks: 9 taps * 256 channels / 32
#define LDSROW 528       // bytes per A row: 512B tap data + 16B pad
                         // (132 dw stride -> b128 frag reads touch every bank
                         //  exactly twice = LDS bandwidth floor)

// ---------------------------------------------------------------------------
// 1) NCHW -> NHWC transpose of x (makes per-pixel channel gathers coalesced).
// ---------------------------------------------------------------------------
__global__ __launch_bounds__(256) void k_transpose(const float* __restrict__ x,
                                                   float* __restrict__ xt) {
  __shared__ float tile[32][33];
  const int b  = blockIdx.z;
  const int c0 = blockIdx.y * 32;
  const int s0 = blockIdx.x * 32;       // s = h*W + w (spatial index)
  const int tx = threadIdx.x, ty = threadIdx.y;
  const float* src = x  + (size_t)b * C_ * HW_;
  float*       dst = xt + (size_t)b * HW_ * C_;
#pragma unroll
  for (int i = 0; i < 4; ++i)
    tile[ty + i * 8][tx] = src[(size_t)(c0 + ty + i * 8) * HW_ + s0 + tx];
  __syncthreads();
#pragma unroll
  for (int i = 0; i < 4; ++i)
    dst[(size_t)(s0 + ty + i * 8) * C_ + c0 + tx] = tile[tx][ty + i * 8];
}

// ---------------------------------------------------------------------------
// 2) Scale predictor: 3x3 conv C=256 -> 2, zero pad, then exp(clip(.,-2,2)).
//    One wave per pixel; sp weights staged in LDS; wave shfl-xor reduction.
// ---------------------------------------------------------------------------
__global__ __launch_bounds__(256) void k_scalepred(const float* __restrict__ xt,
                                                   const float* __restrict__ spw,
                                                   const float* __restrict__ spb,
                                                   float* __restrict__ sx,
                                                   float* __restrict__ sy) {
  __shared__ float wlds[2 * 9 * 256];   // [j][tap][c]
  const int tid = threadIdx.x;
  for (int i = tid; i < 2 * 9 * 256; i += 256) {
    const int jj = i / 2304, r = i % 2304, tap = r >> 8, c = r & 255;
    wlds[i] = spw[(size_t)(jj * 256 + c) * 9 + tap];  // spw is (2,C,3,3)
  }
  __syncthreads();

  const int lane = tid & 31;
  const int pix  = blockIdx.x * 8 + (tid >> 5);
  const int b = pix / HW_, rem = pix % HW_, h = rem / W_, w = rem % W_;

  float a0 = 0.f, a1 = 0.f;
#pragma unroll
  for (int tap = 0; tap < 9; ++tap) {
    const int yy = h + tap / 3 - 1, xx = w + tap % 3 - 1;
    if (yy < 0 || yy >= H_ || xx < 0 || xx >= W_) continue;  // zero pad
    const float* base = xt + ((size_t)(b * H_ + yy) * W_ + xx) * C_;
    const float* wrow = wlds + tap * 256;
#pragma unroll
    for (int u = 0; u < 8; ++u) {
      const int c = (u << 5) + lane;
      const float v = base[c];
      a0 += v * wrow[c];
      a1 += v * wrow[2304 + c];
    }
  }
#pragma unroll
  for (int m = 16; m >= 1; m >>= 1) {
    a0 += __shfl_xor(a0, m, 32);
    a1 += __shfl_xor(a1, m, 32);
  }
  if (lane == 0) {
    sx[pix] = expf(fminf(fmaxf(a0 + spb[0], -2.f), 2.f));
    sy[pix] = expf(fminf(fmaxf(a1 + spb[1], -2.f), 2.f));
  }
}

// ---------------------------------------------------------------------------
// 3) Weight pre-swizzle: fp32 (O,C,3,3) -> f16 in the exact wave32 B-operand
//    image of v_wmma_f32_16x16x32_f16. For outchan group ng, chunk kc:
//      element e of lane L  ==  W[k = kc*32 + e%8 + 8*(L>=16) + 16*(e/8)][n = ng*16+L%16]
//    stored so each lane's 16 halves are 32 contiguous bytes.
// ---------------------------------------------------------------------------
__global__ __launch_bounds__(256) void k_wswizzle(const float* __restrict__ wgt,
                                                  _Float16* __restrict__ bsw) {
  const int idx = blockIdx.x * 256 + threadIdx.x;   // < 16*72*32*16 = 589824
  const int e    = idx & 15;
  int t          = idx >> 4;
  const int lane = t & 31;  t >>= 5;
  const int kc   = t % KC_;
  const int ng   = t / KC_;
  const int k    = kc * 32 + (e & 7) + ((lane >> 4) << 3) + ((e >> 3) << 4);   // GEMM K index
  const int o    = (ng << 4) + (lane & 15);            // outchan (B column)
  const int c    = k & 255, tap = k >> 8;              // k = tap*256 + c
  bsw[idx] = (_Float16)wgt[(size_t)o * 2304 + c * 9 + tap];  // wgt (O,C,9)
}

// ---------------------------------------------------------------------------
// 4) Main fused kernel. Block = 512 threads (16 waves) = 16 pixels x 256 out.
//    Per tap: thread (pixel p, lane) bilinear-samples channels [8*lane,8*lane+8)
//    with 8 coalesced global_load_b128, packs to f16, one ds_store_b128 into
//    the double-buffered tap tile; one barrier; then every wave runs 8 WMMAs
//    (chunks of K=32) against its pre-swizzled B fragments.
// ---------------------------------------------------------------------------
__global__ __launch_bounds__(512) void k_main(const float* __restrict__ xt,
                                              const float* __restrict__ sx,
                                              const float* __restrict__ sy,
                                              const _Float16* __restrict__ bsw,
                                              const float* __restrict__ bias,
                                              float* __restrict__ out) {
  __shared__ __align__(16) unsigned char alds[2][16 * LDSROW];

  const int tid  = threadIdx.x;
  const int p    = tid >> 5;          // pixel-in-tile == wave id == outchan grp
  const int lane = tid & 31;
  const int b = blockIdx.z, h = blockIdx.y, w0 = blockIdx.x << 4;
  const int w = w0 + p;
  const int pix = (b * H_ + h) * W_ + w;
  const float sxv = sx[pix], syv = sy[pix];

  union Frag { v16h v; v8h half8[2]; };
  v8f acc = {};

  for (int tap = 0; tap < 9; ++tap) {
    // tap offset (x = tap%3 - 1, y = tap/3 - 1), scaled per pixel
    const float cx = (float)(tap % 3) - 1.f;
    const float cy = (float)(tap / 3) - 1.f;
    const float gx = fminf(fmaxf((float)w + cx * sxv, 0.f), (float)(W_ - 1));
    const float gy = fminf(fmaxf((float)h + cy * syv, 0.f), (float)(H_ - 1));
    const float x0f = floorf(gx), y0f = floorf(gy);
    const float fx = gx - x0f,  fy = gy - y0f;
    const int ix0 = (int)x0f, iy0 = (int)y0f;
    const int ix1 = min(ix0 + 1, W_ - 1), iy1 = min(iy0 + 1, H_ - 1);
    const float4* q00 = (const float4*)(xt + ((size_t)((b * H_ + iy0) * W_ + ix0)) * C_);
    const float4* q01 = (const float4*)(xt + ((size_t)((b * H_ + iy0) * W_ + ix1)) * C_);
    const float4* q10 = (const float4*)(xt + ((size_t)((b * H_ + iy1) * W_ + ix0)) * C_);
    const float4* q11 = (const float4*)(xt + ((size_t)((b * H_ + iy1) * W_ + ix1)) * C_);
    const float w00 = (1.f - fx) * (1.f - fy), w01 = fx * (1.f - fy);
    const float w10 = (1.f - fx) * fy,         w11 = fx * fy;

    // Sample this thread's 8 channels (c = 8*lane .. 8*lane+7): 8 x b128 loads.
    const int i0 = lane << 1;           // float4 index (c0/4)
    const float4 a00 = q00[i0],     a01 = q01[i0],     a10 = q10[i0],     a11 = q11[i0];
    const float4 b00 = q00[i0 + 1], b01 = q01[i0 + 1], b10 = q10[i0 + 1], b11 = q11[i0 + 1];
    v8h hv;
    hv[0] = (_Float16)(w00 * a00.x + w01 * a01.x + w10 * a10.x + w11 * a11.x);
    hv[1] = (_Float16)(w00 * a00.y + w01 * a01.y + w10 * a10.y + w11 * a11.y);
    hv[2] = (_Float16)(w00 * a00.z + w01 * a01.z + w10 * a10.z + w11 * a11.z);
    hv[3] = (_Float16)(w00 * a00.w + w01 * a01.w + w10 * a10.w + w11 * a11.w);
    hv[4] = (_Float16)(w00 * b00.x + w01 * b01.x + w10 * b10.x + w11 * b11.x);
    hv[5] = (_Float16)(w00 * b00.y + w01 * b01.y + w10 * b10.y + w11 * b11.y);
    hv[6] = (_Float16)(w00 * b00.z + w01 * b01.z + w10 * b10.z + w11 * b11.z);
    hv[7] = (_Float16)(w00 * b00.w + w01 * b01.w + w10 * b10.w + w11 * b11.w);

    const int buf = tap & 1;            // double buffer across taps
    *(v8h*)(&alds[buf][p * LDSROW + (lane << 4)]) = hv;
    __syncthreads();                    // one barrier per tap (9 total)

    // A fragment base: lane holds row M=lane%16; per chunk cc the K quarters
    // sit at bytes cc*64 + {0,32} (+16 for lanes>=16).
    const unsigned char* arow =
        &alds[buf][(lane & 15) * LDSROW + ((lane >> 4) << 4)];
    const _Float16* bbase =
        bsw + ((size_t)((p * KC_ + tap * 8) * 32 + lane) << 4);

#pragma unroll
    for (int cc = 0; cc < 8; ++cc) {
      Frag a, bb;
      a.half8[0] = *(const v8h*)(arow + cc * 64);
      a.half8[1] = *(const v8h*)(arow + cc * 64 + 32);
      const _Float16* bp = bbase + cc * 512;   // next chunk: 32 lanes * 16 halves
      bb.half8[0] = *(const v8h*)(bp);
      bb.half8[1] = *(const v8h*)(bp + 8);
      acc = __builtin_amdgcn_wmma_f32_16x16x32_f16(
          /*neg_a=*/false, a.v, /*neg_b=*/false, bb.v,
          /*c_mod=*/(short)0, acc, /*reuse_a=*/false, /*reuse_b=*/false);
    }
  }

  // Epilogue: D lane = outchan N = p*16 + lane%16; VGPR r -> pixel M = r+8*(lane/16).
  const int o  = (p << 4) + (lane & 15);
  const float bv = bias[o];
  float* op = out + ((size_t)((b * O_ + o) * H_ + h)) * W_ + w0 + ((lane >> 4) << 3);
  const float4 lo = make_float4(acc[0] + bv, acc[1] + bv, acc[2] + bv, acc[3] + bv);
  const float4 hi = make_float4(acc[4] + bv, acc[5] + bv, acc[6] + bv, acc[7] + bv);
  *(float4*)(op)     = lo;
  *(float4*)(op + 4) = hi;
}

// ---------------------------------------------------------------------------
// Launch. Inputs: x, weight, bias, sp_w, sp_b (all fp32). Output (B,O,H,W) f32.
// Workspace layout: xt (151.0MB) | sx (0.56MB) | sy (0.56MB) | bsw (1.13MB).
// ---------------------------------------------------------------------------
extern "C" void kernel_launch(void* const* d_in, const int* in_sizes, int n_in,
                              void* d_out, int out_size, void* d_ws, size_t ws_size,
                              hipStream_t stream) {
  const float* x    = (const float*)d_in[0];
  const float* wgt  = (const float*)d_in[1];
  const float* bias = (const float*)d_in[2];
  const float* spw  = (const float*)d_in[3];
  const float* spb  = (const float*)d_in[4];
  float* out = (float*)d_out;

  char* ws = (char*)d_ws;
  const size_t xt_bytes = (size_t)B_ * HW_ * C_ * sizeof(float);   // 150,994,944
  const size_t s_bytes  = (size_t)B_ * HW_ * sizeof(float);        //     589,824
  float*    xt  = (float*)ws;
  float*    sxp = (float*)(ws + xt_bytes);
  float*    syp = (float*)(ws + xt_bytes + s_bytes);
  _Float16* bsw = (_Float16*)(ws + xt_bytes + 2 * s_bytes);        // 1,179,648 B

  k_transpose<<<dim3(HW_ / 32, C_ / 32, B_), dim3(32, 8), 0, stream>>>(x, xt);
  k_scalepred<<<dim3(B_ * HW_ / 8), dim3(256), 0, stream>>>(xt, spw, spb, sxp, syp);
  k_wswizzle<<<dim3((16 * KC_ * 32 * 16) / 256), dim3(256), 0, stream>>>(wgt, bsw);
  k_main<<<dim3(W_ / 16, H_, B_), dim3(512), 0, stream>>>(xt, sxp, syp, bsw, bias, out);
}
